// TDTreeGRU_90108413870168
// MI455X (gfx1250) — matrix operand order repaired and checked by reading, not compiled, not run on hardware
//
#include <hip/hip_runtime.h>
#include <hip/hip_bf16.h>

typedef __attribute__((ext_vector_type(16))) __bf16 bf16x16;
typedef __attribute__((ext_vector_type(8)))  __bf16 bf16x8;
typedef __attribute__((ext_vector_type(4)))  __bf16 bf16x4;
typedef __attribute__((ext_vector_type(8)))  float  v8f;

#define B_SZ 8
#define DEPTH_ 12
#define NNODES 4095
#define HD 512
#define ID 512
#define G3 1536
#define NK32 16           // K/32 for K=512

// ---------------------------------------------------------------- utilities
__device__ __forceinline__ float fast_sigmoid(float x) {
    return 1.0f / (1.0f + __expf(-x));
}
__device__ __forceinline__ float fast_tanh(float x) {
    float e2 = __expf(2.0f * x);
    return (e2 - 1.0f) / (e2 + 1.0f);
}
__device__ __forceinline__ v8f wmma_bf16(bf16x16 a, bf16x16 b, v8f c) {
    return __builtin_amdgcn_wmma_f32_16x16x32_bf16(false, a, false, b, (short)0, c,
                                                   false, false);
}
__device__ __forceinline__ bf16x16 frag_cat(bf16x8 lo, bf16x8 hi) {
    return __builtin_shufflevector(lo, hi, 0, 1, 2, 3, 4, 5, 6, 7,
                                           8, 9, 10, 11, 12, 13, 14, 15);
}
// A fragment from a row-major bf16 tile (16 x lda): two 16B loads per lane.
__device__ __forceinline__ bf16x16 load_a_frag(const __bf16* base, int lda, int kk, int lane) {
    int m  = lane & 15;
    int kb = (lane < 16) ? 0 : 16;
    const __bf16* p = base + m * lda + kk + kb;
    bf16x8 lo = *(const bf16x8*)p;
    bf16x8 hi = *(const bf16x8*)(p + 8);
    return frag_cat(lo, hi);
}
// B fragment from packed weights: 16 contiguous halves per lane (two 16B loads).
__device__ __forceinline__ bf16x16 load_b_frag(const __bf16* P, int ntile, int kt, int lane) {
    const __bf16* p = P + ((((size_t)ntile * NK32 + kt) * 32 + lane) << 4);
    bf16x8 lo = *(const bf16x8*)p;
    bf16x8 hi = *(const bf16x8*)(p + 8);
    return frag_cat(lo, hi);
}

// ---------------------------------------------------------------- converts
__global__ __launch_bounds__(256) void f32_to_bf16_kernel(const float* __restrict__ src,
                                                          __bf16* __restrict__ dst,
                                                          size_t n) {
    size_t i = (size_t)blockIdx.x * blockDim.x + threadIdx.x;
    size_t stride = (size_t)gridDim.x * blockDim.x;
    for (; i < n; i += stride) dst[i] = (__bf16)src[i];
}

// Pack W (K x N, f32, row-major) into WMMA B-fragment order (bf16):
// P[(((nt*NK32 + kt)*32 + lane)*16 + i] = W[(kt*32 + (lane>=16)*16 + i)*N + nt*16 + (lane&15)]
__global__ __launch_bounds__(256) void pack_w_kernel(const float* __restrict__ W,
                                                     __bf16* __restrict__ P,
                                                     int K, int N) {
    int idx = blockIdx.x * 256 + threadIdx.x;
    if (idx >= K * N) return;
    int i    = idx & 15;
    int lane = (idx >> 4) & 31;
    int blk  = idx >> 9;              // nt*(K/32) + kt
    int nK32 = K >> 5;
    int kt = blk % nK32;
    int nt = blk / nK32;
    int k = kt * 32 + ((lane >> 4) << 4) + i;
    int n = nt * 16 + (lane & 15);
    P[idx] = (__bf16)W[(size_t)k * N + n];
}

// ---------------------------------------------------------------- big GEMM
// C[M x Nn] (bf16) = A[M x K=512] (bf16) @ Wp (packed bf16) + bias (f32)
// WG: 256 thr = 8 waves; tile 16(M) x 256(N): each wave owns two 16x16 tiles.
__global__ __launch_bounds__(256) void gemm_bias_bf16(const __bf16* __restrict__ A,
                                                      const __bf16* __restrict__ Wp,
                                                      const float* __restrict__ bias,
                                                      __bf16* __restrict__ C,
                                                      int M, int Nn) {
    __shared__ __align__(16) __bf16 At[16 * 512];   // full-K A strip (16 KB)
    int t = threadIdx.x, lane = t & 31, wave = t >> 5;
    int m0 = blockIdx.y * 16;
    int tbase = blockIdx.x * 16;       // ntile base (16 tiles per WG)

    { // stage A strip: 32 elems (4 x b128) per thread
        int e0 = t * 32;
        int row = e0 >> 9, col = e0 & 511;
        int gm = m0 + row;
        bf16x8* dst = (bf16x8*)(At + row * 512 + col);
        if (gm < M) {
            const bf16x8* src = (const bf16x8*)(A + (size_t)gm * ID + col);
            dst[0] = src[0]; dst[1] = src[1]; dst[2] = src[2]; dst[3] = src[3];
        } else {
            bf16x8 z = {};
            dst[0] = z; dst[1] = z; dst[2] = z; dst[3] = z;
        }
    }
    __syncthreads();

    int t0 = tbase + wave;            // first N-tile
    int t1 = tbase + 8 + wave;        // second N-tile
    v8f acc0 = {}, acc1 = {};
#pragma unroll
    for (int kt = 0; kt < NK32; ++kt) {
        bf16x16 af = load_a_frag(At, 512, kt * 32, lane);
        bf16x16 b0 = load_b_frag(Wp, t0, kt, lane);
        bf16x16 b1 = load_b_frag(Wp, t1, kt, lane);
        acc0 = wmma_bf16(af, b0, acc0);
        acc1 = wmma_bf16(af, b1, acc1);
    }

    int mb = (lane < 16) ? 0 : 8;
    int n0 = t0 * 16 + (lane & 15);
    int n1 = t1 * 16 + (lane & 15);
    float bv0 = bias[n0], bv1 = bias[n1];
#pragma unroll
    for (int r = 0; r < 8; ++r) {
        int gm = m0 + mb + r;
        if (gm < M) {
            C[(size_t)gm * Nn + n0] = (__bf16)(acc0[r] + bv0);
            C[(size_t)gm * Nn + n1] = (__bf16)(acc1[r] + bv1);
        }
    }
}

// ---------------------------------------------------------------- root node
__global__ __launch_bounds__(256) void root_kernel(const __bf16* __restrict__ gx,
                                                   const __bf16* __restrict__ cx,
                                                   float* __restrict__ h,
                                                   float* __restrict__ out) {
    int e = blockIdx.x * blockDim.x + threadIdx.x;
    if (e >= B_SZ * HD) return;
    int b = e >> 9, col = e & (HD - 1);
    size_t mrow = (size_t)b * NNODES;   // node 0
    float z    = fast_sigmoid((float)gx[mrow * G3 + 2 * HD + col]);
    float cell = fast_tanh((float)cx[mrow * HD + col]);
    float hv = z * cell;                 // lh = rh = 0 at root
    h[mrow * HD + col] = hv;
    out[((size_t)b * NNODES + (NNODES - 1)) * HD + col] = hv;      // rotated slot
    out[(size_t)B_SZ * NNODES * HD + (size_t)b * HD + col] = hv;   // second output
}

// ---------------------------------------------------------------- tree level
// One WG = 16 rows of (b,node) for one child-side of one level.
// node = set_base + 2*set_idx ; row r: b = r%8, set_idx = r/8.
__global__ __launch_bounds__(256) void tree_level_kernel(const __bf16* __restrict__ Wgp, // packed 512x1536
                                                         const __bf16* __restrict__ Wcp, // packed 512x512
                                                         const __bf16* __restrict__ gx,  // (B*N)x1536
                                                         const __bf16* __restrict__ cx,  // (B*N)x512
                                                         float* __restrict__ h,          // (B*N)x512
                                                         float* __restrict__ out,
                                                         int set_base, int count) {
    __shared__ __align__(16) float  ph [16 * HD];   // parent h f32        (32 KB)
    __shared__ __align__(16) __bf16 phb[16 * HD];   // parent h bf16       (16 KB)
    __shared__ __align__(16) __bf16 a2 [16 * HD];   // rp*ph bf16 (A of G2)(16 KB)
    __shared__ __align__(16) float  zp [16 * HD];   // gate zp             (32 KB)
    __shared__ __align__(16) float  zg [16 * HD];   // gate z              (32 KB)

    int t = threadIdx.x, lane = t & 31, wave = t >> 5;
    int r0 = blockIdx.x * 16;

    // ---- phase 1: gather parent-h rows into LDS (f32 + bf16 mirror)
#pragma unroll
    for (int i = 0; i < 8; ++i) {
        int e4 = t + i * 256;               // 16*128 float4 slots
        int row = e4 >> 7;
        int c4 = e4 & 127;
        int gr = r0 + row;
        int b = gr & 7, si = gr >> 3;
        float4 v = make_float4(0.f, 0.f, 0.f, 0.f);
        if (si < count) {
            int node = set_base + 2 * si;
            int par = (node - 1) >> 1;
            v = *(const float4*)(h + ((size_t)b * NNODES + par) * HD + c4 * 4);
        }
        *(float4*)(ph + row * HD + c4 * 4) = v;
        bf16x4 vb;
        vb[0] = (__bf16)v.x; vb[1] = (__bf16)v.y;
        vb[2] = (__bf16)v.z; vb[3] = (__bf16)v.w;
        *(bf16x4*)(phb + row * HD + c4 * 4) = vb;
    }
    __syncthreads();

    int nl = lane & 15;
    int mb = (lane < 16) ? 0 : 8;

    // ---- phase 2: gates = sigmoid(gx + ph @ Wg) ; produce a2 / zp / z
    for (int nt = wave; nt < 96; nt += 8) {
        v8f acc = {};
#pragma unroll
        for (int kt = 0; kt < NK32; ++kt) {
            bf16x16 af  = load_a_frag(phb, HD, kt * 32, lane);
            bf16x16 bfr = load_b_frag(Wgp, nt, kt, lane);
            acc = wmma_bf16(af, bfr, acc);
        }
        int col = nt * 16 + nl;
#pragma unroll
        for (int r = 0; r < 8; ++r) {
            int m = mb + r;
            int gr = r0 + m;
            int b = gr & 7, si = gr >> 3;
            float gxv = 0.f;
            if (si < count) {
                int node = set_base + 2 * si;
                gxv = (float)gx[((size_t)b * NNODES + node) * G3 + col];
            }
            float g = fast_sigmoid(acc[r] + gxv);
            if (col < HD)           a2[m * HD + col]            = (__bf16)(g * ph[m * HD + col]);
            else if (col < 2 * HD)  zp[m * HD + (col - HD)]     = g;
            else                    zg[m * HD + (col - 2 * HD)] = g;
        }
    }
    __syncthreads();

    // ---- phase 3: cell = tanh(cx + a2 @ Wc) ; h = zp*ph + z*cell
    for (int nt = wave; nt < 32; nt += 8) {
        v8f acc = {};
#pragma unroll
        for (int kt = 0; kt < NK32; ++kt) {
            bf16x16 af  = load_a_frag(a2, HD, kt * 32, lane);
            bf16x16 bfr = load_b_frag(Wcp, nt, kt, lane);
            acc = wmma_bf16(af, bfr, acc);
        }
        int col = nt * 16 + nl;
#pragma unroll
        for (int r = 0; r < 8; ++r) {
            int m = mb + r;
            int gr = r0 + m;
            int b = gr & 7, si = gr >> 3;
            if (si >= count) continue;
            int node = set_base + 2 * si;
            float cxv = (float)cx[((size_t)b * NNODES + node) * HD + col];
            float cell = fast_tanh(acc[r] + cxv);
            float hv = zp[m * HD + col] * ph[m * HD + col] + zg[m * HD + col] * cell;
            h[((size_t)b * NNODES + node) * HD + col] = hv;
            out[((size_t)b * NNODES + (node - 1)) * HD + col] = hv;   // rotate-by-one
        }
    }
}

// ---------------------------------------------------------------- launcher
extern "C" void kernel_launch(void* const* d_in, const int* in_sizes, int n_in,
                              void* d_out, int out_size, void* d_ws, size_t ws_size,
                              hipStream_t stream) {
    (void)in_sizes; (void)n_in; (void)out_size; (void)ws_size;
    const int M = B_SZ * NNODES;          // 32760

    const float* inputs = (const float*)d_in[0];
    const float* W_gih  = (const float*)d_in[2];
    const float* b_gih  = (const float*)d_in[3];
    const float* W_glhh = (const float*)d_in[4];
    const float* W_grhh = (const float*)d_in[5];
    const float* W_cih  = (const float*)d_in[6];
    const float* b_cih  = (const float*)d_in[7];
    const float* W_clhh = (const float*)d_in[8];
    const float* W_crhh = (const float*)d_in[9];
    float* out = (float*)d_out;

    // workspace carve-out (256B aligned)
    char* p = (char*)d_ws;
    auto carve = [&](size_t bytes) {
        char* q = p;
        p += (bytes + 255) & ~(size_t)255;
        return q;
    };
    __bf16* inputs_bf = (__bf16*)carve((size_t)M * ID * 2);
    __bf16* Wgih_p    = (__bf16*)carve((size_t)ID * G3 * 2);
    __bf16* Wglhh_p   = (__bf16*)carve((size_t)HD * G3 * 2);
    __bf16* Wgrhh_p   = (__bf16*)carve((size_t)HD * G3 * 2);
    __bf16* Wcih_p    = (__bf16*)carve((size_t)ID * HD * 2);
    __bf16* Wclhh_p   = (__bf16*)carve((size_t)HD * HD * 2);
    __bf16* Wcrhh_p   = (__bf16*)carve((size_t)HD * HD * 2);
    __bf16* gx_bf     = (__bf16*)carve((size_t)M * G3 * 2);
    __bf16* cx_bf     = (__bf16*)carve((size_t)M * HD * 2);
    float*  h_f       = (float*) carve((size_t)M * HD * 4);

    // 1) activation conversion + weight packing into fragment order
    f32_to_bf16_kernel<<<4096, 256, 0, stream>>>(inputs, inputs_bf, (size_t)M * ID);
    pack_w_kernel<<<(ID * G3 + 255) / 256, 256, 0, stream>>>(W_gih,  Wgih_p,  ID, G3);
    pack_w_kernel<<<(HD * G3 + 255) / 256, 256, 0, stream>>>(W_glhh, Wglhh_p, HD, G3);
    pack_w_kernel<<<(HD * G3 + 255) / 256, 256, 0, stream>>>(W_grhh, Wgrhh_p, HD, G3);
    pack_w_kernel<<<(ID * HD + 255) / 256, 256, 0, stream>>>(W_cih,  Wcih_p,  ID, HD);
    pack_w_kernel<<<(HD * HD + 255) / 256, 256, 0, stream>>>(W_clhh, Wclhh_p, HD, HD);
    pack_w_kernel<<<(HD * HD + 255) / 256, 256, 0, stream>>>(W_crhh, Wcrhh_p, HD, HD);

    // 2) input projections (parallel GEMMs)
    {
        dim3 g1(G3 / 256, (M + 15) / 16);
        gemm_bias_bf16<<<g1, 256, 0, stream>>>(inputs_bf, Wgih_p, b_gih, gx_bf, M, G3);
        dim3 g2(HD / 256, (M + 15) / 16);
        gemm_bias_bf16<<<g2, 256, 0, stream>>>(inputs_bf, Wcih_p, b_cih, cx_bf, M, HD);
    }

    // 3) root
    root_kernel<<<(B_SZ * HD + 255) / 256, 256, 0, stream>>>(gx_bf, cx_bf, h_f, out);

    // 4) levels 1..11: left children (odd nodes) then right children (even nodes)
    for (int d = 1; d < DEPTH_; ++d) {
        int s   = (1 << d) - 1;       // level start (odd -> left child)
        int cnt = 1 << (d - 1);       // nodes per side
        int wgs = (B_SZ * cnt + 15) / 16;
        tree_level_kernel<<<wgs, 256, 0, stream>>>(Wglhh_p, Wclhh_p, gx_bf, cx_bf,
                                                   h_f, out, s, cnt);
        tree_level_kernel<<<wgs, 256, 0, stream>>>(Wgrhh_p, Wcrhh_p, gx_bf, cx_bf,
                                                   h_f, out, s + 1, cnt);
    }
}
